// PDG2Seq_22969485099155
// MI455X (gfx1250) — matrix operand order repaired and checked by reading, not compiled, not run on hardware
//
#include <hip/hip_runtime.h>
#include <hip/hip_bf16.h>
#include <math.h>

// Problem constants (from reference setup_inputs)
#define NB   8      // batch
#define TT   12     // time steps
#define NN   1024   // nodes
#define EMB  16
#define HID  64
#define DINH 97     // DIN + HID = 33 + 64
#define KPAD 128    // DINH padded to WMMA-friendly 128

typedef __bf16 bf16;
typedef __attribute__((ext_vector_type(16))) __bf16 v16bf;
typedef __attribute__((ext_vector_type(8)))  float  v8f;

// ---------------------------------------------------------------------------
// WMMA fragment helpers (layouts per CDNA5 ISA 7.12.2, wave32)
// A-matrix 16x32 bf16: lane&15 = M row; element e -> K = e + (e>=8?8:0) + (lane>=16?8:0)
// B-matrix 32x16 bf16: lane&15 = N col; element e -> K = e + (e>=8?8:0) + (lane>=16?8:0)
// C/D 16x16 f32: VGPR r -> M = r + (lane>=16?8:0); N = lane&15
// ---------------------------------------------------------------------------
__device__ __forceinline__ v16bf load_a_frag(const bf16* p, int ld, int lane) {
  int row  = lane & 15;
  int koff = (lane & 16) ? 8 : 0;
  const bf16* r = p + (size_t)row * ld;
  v16bf a;
#pragma unroll
  for (int e = 0; e < 16; ++e) {
    int k = e + ((e >= 8) ? 8 : 0) + koff;
    a[e] = r[k];
  }
  return a;
}

__device__ __forceinline__ v8f wmma_bf16(v16bf a, v16bf b, v8f c) {
  return __builtin_amdgcn_wmma_f32_16x16x32_bf16(false, a, false, b, (short)0, c,
                                                 false, false);
}

// ---------------------------------------------------------------------------
// K0: temporal embedding gathers (tiny)
// ---------------------------------------------------------------------------
__global__ void embed_kernel(const int* __restrict__ tod_idx, const int* __restrict__ dow_idx,
                             const float* __restrict__ TiD1, const float* __restrict__ TiD2,
                             const float* __restrict__ DiW1, const float* __restrict__ DiW2,
                             const float* __restrict__ fc_tod_W, const float* __restrict__ fc_tod_b,
                             const float* __restrict__ fc_dow_W, const float* __restrict__ fc_dow_b,
                             float* __restrict__ todF, float* __restrict__ dowF,
                             float* __restrict__ e1v, float* __restrict__ e2v) {
  int idx = blockIdx.x * blockDim.x + threadIdx.x;
  if (idx >= NB * TT * EMB) return;
  int j  = idx & (EMB - 1);
  int bt = idx >> 4;            // b*TT + t
  int tod = tod_idx[bt];
  int dow = dow_idx[bt];
  todF[idx] = fc_tod_W[tod * EMB + j] + fc_tod_b[j];
  dowF[idx] = fc_dow_W[dow * EMB + j] + fc_dow_b[j];
  e1v[idx]  = TiD1[tod * EMB + j] * DiW1[dow * EMB + j];
  e2v[idx]  = TiD2[tod * EMB + j] * DiW2[dow * EMB + j];
}

// ---------------------------------------------------------------------------
// K1: hoisted node-adaptive weights (time-invariant), stored in WMMA
// B-fragment-native layout: [n][kblock(8)][tc(Do/16)][lane(32)][e(16)] bf16,
// so node_gemm_wmma reads 16 contiguous bf16 per lane (2x global_load_b128).
// Source semantics: W[n][kcheb][i][o] = sum_d node_emb[n,d]*pool[d][kcheb][i][o]
// composite K row = kcheb*128 + i (i zero-padded 97->128)
// ---------------------------------------------------------------------------
__global__ void node_weight_kernel(const float* __restrict__ emb, const float* __restrict__ pool,
                                   bf16* __restrict__ out, int Do) {
  long long idx = (long long)blockIdx.x * blockDim.x + threadIdx.x;
  long long total = (long long)NN * 2 * KPAD * Do;
  if (idx >= total) return;
  int o = (int)(idx % Do);
  long long t = idx / Do;
  int i = (int)(t % KPAD); t /= KPAD;
  int kcheb = (int)(t % 2);
  int n = (int)(t / 2);
  float v = 0.f;
  if (i < DINH) {
#pragma unroll
    for (int d = 0; d < EMB; ++d)
      v += emb[n * EMB + d] * pool[(((size_t)d * 2 + kcheb) * DINH + i) * Do + o];
  }
  // scatter into fragment-native layout
  int k    = kcheb * KPAD + i;        // composite K row, 0..255
  int kbk  = k >> 5;                  // k-block, 0..7
  int k5   = k & 31;
  int lane = (((k5 >> 3) & 1) << 4) | (o & 15);
  int e    = (k5 & 7) | ((k5 >> 4) << 3);
  int tc   = o >> 4;
  int tpw  = Do >> 4;
  size_t didx = ((((size_t)n * 8 + kbk) * tpw + tc) * 32 + lane) * (size_t)16 + e;
  out[didx] = (bf16)v;
}

__global__ void node_bias_kernel(const float* __restrict__ emb, const float* __restrict__ bpool,
                                 float* __restrict__ out, int Do) {
  int idx = blockIdx.x * blockDim.x + threadIdx.x;
  if (idx >= NN * Do) return;
  int o = idx % Do, n = idx / Do;
  float v = 0.f;
#pragma unroll
  for (int d = 0; d < EMB; ++d) v += emb[n * EMB + d] * bpool[d * Do + o];
  out[idx] = v;
}

// ---------------------------------------------------------------------------
// K2: E1/E2 = tanh(node_emb * e{1,2}[b,t]) -> bf16 [B,N,32] (K zero-padded)
// ---------------------------------------------------------------------------
__global__ void e_embed_kernel(const float* __restrict__ node_emb,
                               const float* __restrict__ e1v, const float* __restrict__ e2v,
                               int t, bf16* __restrict__ E1b, bf16* __restrict__ E2b) {
  int idx = blockIdx.x * blockDim.x + threadIdx.x;   // (b,n,k) k<32
  if (idx >= NB * NN * 32) return;
  int k = idx & 31;
  int n = (idx >> 5) & (NN - 1);
  int b = idx >> 15;
  bf16 v1 = (bf16)0.f, v2 = (bf16)0.f;
  if (k < EMB) {
    float ne = node_emb[n * EMB + k];
    v1 = (bf16)tanhf(ne * e1v[(b * TT + t) * EMB + k]);
    v2 = (bf16)tanhf(ne * e2v[(b * TT + t) * EMB + k]);
  }
  E1b[idx] = v1;
  E2b[idx] = v2;
}

// ---------------------------------------------------------------------------
// K3: adjacency logits L = relu(E1 @ E2^T) per batch, one WMMA per tile.
// ---------------------------------------------------------------------------
__global__ void adj_logits_wmma(const bf16* __restrict__ E1b, const bf16* __restrict__ E2b,
                                float* __restrict__ logits) {
  int wave = (blockIdx.x * blockDim.x + threadIdx.x) >> 5;
  int lane = threadIdx.x & 31;
  int b    = wave >> 12;          // 4096 tiles / batch
  int tile = wave & 4095;
  int tr = tile >> 6, tc = tile & 63;

  const bf16* e1p = E1b + ((size_t)b * NN + tr * 16) * 32;
  const bf16* e2p = E2b + ((size_t)b * NN + tc * 16) * 32;

  v16bf a = load_a_frag(e1p, 32, lane);
  int col = lane & 15;
  int kb  = (lane & 16) ? 8 : 0;
  v16bf bb;
#pragma unroll
  for (int e = 0; e < 16; ++e) bb[e] = e2p[(size_t)col * 32 + kb + e + ((e >= 8) ? 8 : 0)];

  v8f c;
#pragma unroll
  for (int r = 0; r < 8; ++r) c[r] = 0.f;
  c = wmma_bf16(a, bb, c);

  float* out = logits + (size_t)b * NN * NN + (size_t)(tr * 16) * NN + tc * 16;
  int m0 = (lane & 16) ? 8 : 0;
#pragma unroll
  for (int r = 0; r < 8; ++r) {
    float v = c[r];
    out[(size_t)(m0 + r) * NN + col] = v > 0.f ? v : 0.f;   // relu fused
  }
}

// ---------------------------------------------------------------------------
// K4: row softmax over N=1024 -> bf16 adjacency (16 MB, L2-resident)
// ---------------------------------------------------------------------------
__global__ void softmax_rows(const float* __restrict__ logits, bf16* __restrict__ Ab) {
  __shared__ float red[256];
  int row = blockIdx.x;               // b*NN + n
  int tid = threadIdx.x;
  const float* rp = logits + (size_t)row * NN;

  float m = -1e30f;
  for (int j = tid; j < NN; j += 256) m = fmaxf(m, rp[j]);
  red[tid] = m; __syncthreads();
  for (int s = 128; s > 0; s >>= 1) { if (tid < s) red[tid] = fmaxf(red[tid], red[tid + s]); __syncthreads(); }
  float mx = red[0]; __syncthreads();

  float sum = 0.f;
  for (int j = tid; j < NN; j += 256) sum += expf(rp[j] - mx);
  red[tid] = sum; __syncthreads();
  for (int s = 128; s > 0; s >>= 1) { if (tid < s) red[tid] += red[tid + s]; __syncthreads(); }
  float inv = 1.f / red[0];

  bf16* ap = Ab + (size_t)row * NN;
  for (int j = tid; j < NN; j += 256) ap[j] = (bf16)(expf(rp[j] - mx) * inv);
}

// ---------------------------------------------------------------------------
// K5: build bf16 cell input xg[b,n,0:128] = [x | todF | dowF | (z*)state | 0pad]
// ---------------------------------------------------------------------------
__global__ void build_xg(const float* __restrict__ x_data, const float* __restrict__ todF,
                         const float* __restrict__ dowF, const float* __restrict__ state,
                         const float* __restrict__ zr, int use_z, int t,
                         bf16* __restrict__ xg) {
  int idx = blockIdx.x * blockDim.x + threadIdx.x;   // (b,n,j) j<128
  if (idx >= NB * NN * KPAD) return;
  int j = idx & (KPAD - 1);
  int n = (idx >> 7) & (NN - 1);
  int b = idx >> 17;
  float v;
  if (j == 0)       v = x_data[(size_t)(b * TT + t) * NN + n];
  else if (j < 17)  v = todF[(b * TT + t) * EMB + (j - 1)];
  else if (j < 33)  v = dowF[(b * TT + t) * EMB + (j - 17)];
  else if (j < DINH) {
    int h = j - 33;
    float s = state[((size_t)b * NN + n) * HID + h];
    if (use_z) s *= zr[((size_t)b * NN + n) * 128 + h];   // z gate
    v = s;
  } else v = 0.f;
  xg[idx] = (bf16)v;
}

// ---------------------------------------------------------------------------
// K6: x1 = A @ xg  per batch: [1024,1024]x[1024,128] bf16 WMMA.
// Block-cooperative: one 256-thread block (8 waves) owns a 16x128 output slab
// (block = b*64 + M-tile). Per 32-deep k-step the contiguous 32x128 bf16 slab
// of xg is staged into LDS with CDNA5 async copies (global_load_async_to_lds_
// b128, ASYNCcnt), then each wave runs one WMMA against its 16-col B tile.
// A-fragment rows come straight from the L2-resident bf16 adjacency.
// ---------------------------------------------------------------------------
__global__ void __launch_bounds__(256) ax_wmma(const bf16* __restrict__ Ab,
                                               const bf16* __restrict__ xg,
                                               bf16* __restrict__ x1) {
  __shared__ bf16 sh[32 * KPAD];                 // 8 KB k-slab
  int tid  = threadIdx.x;
  int wave = tid >> 5;                           // N-tile (0..7)
  int lane = tid & 31;
  int blk  = blockIdx.x;                         // b*64 + tr
  int b  = blk >> 6;
  int tr = blk & 63;

  const bf16* Ap = Ab + (size_t)b * NN * NN + (size_t)(tr * 16) * NN;
  const bf16* Xp = xg + (size_t)b * NN * KPAD;

  int col = lane & 15;
  int kb  = (lane & 16) ? 8 : 0;
  v8f c;
#pragma unroll
  for (int r = 0; r < 8; ++r) c[r] = 0.f;

  for (int kk = 0; kk < NN; kk += 32) {
    // ---- async-stage the contiguous 32x128 xg slab into LDS (2 x 512B/wave)
#pragma unroll
    for (int it = 0; it < 2; ++it) {
      int elem = (it * 256 + tid) * 8;           // 8 bf16 = 16 B per lane
      unsigned ldsoff = (unsigned)(size_t)(sh + elem);
      unsigned long long gaddr =
          (unsigned long long)(Xp + (size_t)kk * KPAD + elem);
      asm volatile("global_load_async_to_lds_b128 %0, %1, off"
                   :: "v"(ldsoff), "v"(gaddr) : "memory");
    }
    asm volatile("s_wait_asynccnt 0x0" ::: "memory");
    __syncthreads();

    v16bf a = load_a_frag(Ap + kk, NN, lane);
    v16bf bb;
#pragma unroll
    for (int e = 0; e < 16; ++e) bb[e] = sh[(size_t)(kb + e + ((e >= 8) ? 8 : 0)) * KPAD + wave * 16 + col];
    c = wmma_bf16(a, bb, c);
    __syncthreads();                             // slab reused next k-step
  }

  bf16* out = x1 + (size_t)b * NN * KPAD + (size_t)(tr * 16) * KPAD + wave * 16;
  int m0 = (lane & 16) ? 8 : 0;
#pragma unroll
  for (int r = 0; r < 8; ++r) out[(size_t)(m0 + r) * KPAD + col] = (bf16)c[r];
}

// ---------------------------------------------------------------------------
// K7: batched per-node GEMM: y[b,o] = sum_k [xg;x1][b,n,k] * W[n,k,o] + bias
// M=16 (B=8 padded), composite K=256 (fully unrolled: 8 WMMAs), N-tile=16.
// W is in fragment-native layout (16 contiguous bf16 per lane per k-block).
// mode 0 -> sigmoid (gate zr, Do=128); mode 1 -> tanh (candidate hc, Do=64)
// ---------------------------------------------------------------------------
__global__ void node_gemm_wmma(const bf16* __restrict__ xg, const bf16* __restrict__ x1,
                               const bf16* __restrict__ WnT, const float* __restrict__ bn,
                               float* __restrict__ out, int Do, int mode) {
  int wave = (blockIdx.x * blockDim.x + threadIdx.x) >> 5;
  int lane = threadIdx.x & 31;
  int tpw = Do >> 4;
  int n  = wave / tpw;
  int tc = wave % tpw;

  int row  = lane & 15;                 // batch index (pad to 16)
  int koff = (lane & 16) ? 8 : 0;
  int col  = lane & 15;

  v8f c;
#pragma unroll
  for (int r = 0; r < 8; ++r) c[r] = 0.f;

#pragma unroll
  for (int kbk = 0; kbk < 8; ++kbk) {
    int kk = kbk * 32;
    v16bf a;
#pragma unroll
    for (int e = 0; e < 16; ++e) {
      int k = kk + e + ((e >= 8) ? 8 : 0) + koff;
      bf16 v = (bf16)0.f;
      if (row < NB) {
        if (k < KPAD) v = xg[((size_t)row * NN + n) * KPAD + k];
        else          v = x1[((size_t)row * NN + n) * KPAD + (k - KPAD)];
      }
      a[e] = v;
    }
    const bf16* Wf = WnT + ((((size_t)n * 8 + kbk) * tpw + tc) * 32 + lane) * (size_t)16;
    v16bf bb;
#pragma unroll
    for (int e = 0; e < 16; ++e) bb[e] = Wf[e];
    c = wmma_bf16(a, bb, c);
  }

  int m0 = (lane & 16) ? 8 : 0;
#pragma unroll
  for (int r = 0; r < 8; ++r) {
    int bi = m0 + r;
    if (bi < NB) {
      float v = c[r] + bn[n * Do + tc * 16 + col];
      v = (mode == 0) ? (1.f / (1.f + expf(-v))) : tanhf(v);
      out[((size_t)bi * NN + n) * Do + tc * 16 + col] = v;
    }
  }
}

// ---------------------------------------------------------------------------
// K8: zero init state
// ---------------------------------------------------------------------------
__global__ void zero_state(float* __restrict__ state) {
  int idx = blockIdx.x * blockDim.x + threadIdx.x;
  if (idx < NB * NN * HID) state[idx] = 0.f;
}

// ---------------------------------------------------------------------------
// K9: GRU state update: s = r*s + (1-r)*hc
// ---------------------------------------------------------------------------
__global__ void state_update(const float* __restrict__ zr, const float* __restrict__ hc,
                             float* __restrict__ state, float* __restrict__ statesAll, int t) {
  int idx = blockIdx.x * blockDim.x + threadIdx.x;
  if (idx >= NB * NN * HID) return;
  int h = idx & (HID - 1);
  int n = (idx >> 6) & (NN - 1);
  int b = idx >> 16;
  float r = zr[((size_t)b * NN + n) * 128 + HID + h];
  float s = state[idx];
  float ns = r * s + (1.f - r) * hc[idx];
  state[idx] = ns;
  statesAll[((size_t)(b * TT + t) * NN + n) * HID + h] = ns;
}

// ---------------------------------------------------------------------------
// K10: temporal self-attention per (b,n) over T=12 (tiny; VALU)
// ---------------------------------------------------------------------------
__global__ void attention_kernel(const float* __restrict__ statesAll,
                                 const float* __restrict__ WQ, const float* __restrict__ bQ,
                                 const float* __restrict__ WK, const float* __restrict__ bK,
                                 const float* __restrict__ WV, const float* __restrict__ bV,
                                 float* __restrict__ out) {
  __shared__ float xr[TT][HID], Qs[TT][HID], Ks[TT][HID], Vs[TT][HID], at[TT][TT];
  int bn = blockIdx.x;            // b*NN + n
  int b = bn >> 10, n = bn & (NN - 1);
  int o = threadIdx.x;            // 0..63

  for (int t = 0; t < TT; ++t)
    xr[t][o] = statesAll[((size_t)(b * TT + t) * NN + n) * HID + o];
  __syncthreads();

  for (int t = 0; t < TT; ++t) {
    float q = bQ[o], k = bK[o], v = bV[o];
    for (int h = 0; h < HID; ++h) {
      float x = xr[t][h];
      q += x * WQ[h * HID + o];
      k += x * WK[h * HID + o];
      v += x * WV[h * HID + o];
    }
    Qs[t][o] = q; Ks[t][o] = k; Vs[t][o] = v;
  }
  __syncthreads();

  if (o < TT) {
    float sc[TT], mx = -1e30f;
    for (int s = 0; s < TT; ++s) {
      float d = 0.f;
      for (int h = 0; h < HID; ++h) d += Qs[o][h] * Ks[s][h];
      sc[s] = d * 0.125f;                // 1/sqrt(64)
      mx = fmaxf(mx, sc[s]);
    }
    float sum = 0.f;
    for (int s = 0; s < TT; ++s) { sc[s] = expf(sc[s] - mx); sum += sc[s]; }
    float inv = 1.f / sum;
    for (int s = 0; s < TT; ++s) at[o][s] = sc[s] * inv;
  }
  __syncthreads();

  for (int t = 0; t < TT; ++t) {
    float z = 0.f;
    for (int s = 0; s < TT; ++s) z += at[t][s] * Vs[s][o];
    out[((size_t)(b * TT + t) * NN + n) * HID + o] = z;
  }
}

// ---------------------------------------------------------------------------
// host launcher
// ---------------------------------------------------------------------------
extern "C" void kernel_launch(void* const* d_in, const int* in_sizes, int n_in,
                              void* d_out, int out_size, void* d_ws, size_t ws_size,
                              hipStream_t stream) {
  const float* x_data   = (const float*)d_in[0];
  const int*   tod_idx  = (const int*)  d_in[1];
  const int*   dow_idx  = (const int*)  d_in[2];
  const float* TiD1     = (const float*)d_in[3];
  const float* TiD2     = (const float*)d_in[4];
  const float* DiW1     = (const float*)d_in[5];
  const float* DiW2     = (const float*)d_in[6];
  const float* node_emb = (const float*)d_in[7];
  const float* fc_tod_W = (const float*)d_in[8];
  const float* fc_tod_b = (const float*)d_in[9];
  const float* fc_dow_W = (const float*)d_in[10];
  const float* fc_dow_b = (const float*)d_in[11];
  const float* Wg_pool  = (const float*)d_in[12];
  const float* bg_pool  = (const float*)d_in[13];
  const float* Wu_pool  = (const float*)d_in[14];
  const float* bu_pool  = (const float*)d_in[15];
  const float* WQ = (const float*)d_in[16]; const float* bQ = (const float*)d_in[17];
  const float* WK = (const float*)d_in[18]; const float* bK = (const float*)d_in[19];
  const float* WV = (const float*)d_in[20]; const float* bV = (const float*)d_in[21];
  float* out = (float*)d_out;

  char* w = (char*)d_ws;
  size_t off = 0;
  auto take = [&](size_t bytes) -> void* {
    void* p = (void*)(w + off);
    off += (bytes + 255) & ~(size_t)255;
    return p;
  };
  float* todF = (float*)take((size_t)NB * TT * EMB * 4);
  float* dowF = (float*)take((size_t)NB * TT * EMB * 4);
  float* e1v  = (float*)take((size_t)NB * TT * EMB * 4);
  float* e2v  = (float*)take((size_t)NB * TT * EMB * 4);
  bf16* WgN = (bf16*)take((size_t)NN * 256 * 128 * 2);   // 64 MB, frag-native
  bf16* WuN = (bf16*)take((size_t)NN * 256 * 64 * 2);    // 32 MB, frag-native
  float* bgN = (float*)take((size_t)NN * 128 * 4);
  float* buN = (float*)take((size_t)NN * 64 * 4);
  bf16* E1b = (bf16*)take((size_t)NB * NN * 32 * 2);
  bf16* E2b = (bf16*)take((size_t)NB * NN * 32 * 2);
  float* logits = (float*)take((size_t)NB * NN * NN * 4); // 32 MB
  bf16*  Ab     = (bf16*) take((size_t)NB * NN * NN * 2); // 16 MB (L2-resident)
  bf16*  xg     = (bf16*) take((size_t)NB * NN * KPAD * 2);
  bf16*  x1     = (bf16*) take((size_t)NB * NN * KPAD * 2);
  float* zr     = (float*)take((size_t)NB * NN * 128 * 4);
  float* hc     = (float*)take((size_t)NB * NN * HID * 4);
  float* state  = (float*)take((size_t)NB * NN * HID * 4);
  float* statesAll = (float*)take((size_t)NB * TT * NN * HID * 4); // 24 MB

  // ---- one-time precompute (hoisted out of the recurrent scan) ----
  embed_kernel<<<(NB * TT * EMB + 255) / 256, 256, 0, stream>>>(
      tod_idx, dow_idx, TiD1, TiD2, DiW1, DiW2,
      fc_tod_W, fc_tod_b, fc_dow_W, fc_dow_b, todF, dowF, e1v, e2v);

  {
    long long tg = (long long)NN * 2 * KPAD * 128;
    node_weight_kernel<<<(unsigned)((tg + 255) / 256), 256, 0, stream>>>(node_emb, Wg_pool, WgN, 128);
    long long tu = (long long)NN * 2 * KPAD * 64;
    node_weight_kernel<<<(unsigned)((tu + 255) / 256), 256, 0, stream>>>(node_emb, Wu_pool, WuN, 64);
    node_bias_kernel<<<(NN * 128 + 255) / 256, 256, 0, stream>>>(node_emb, bg_pool, bgN, 128);
    node_bias_kernel<<<(NN * 64 + 255) / 256, 256, 0, stream>>>(node_emb, bu_pool, buN, 64);
  }
  zero_state<<<(NB * NN * HID + 255) / 256, 256, 0, stream>>>(state);

  // ---- recurrent time loop ----
  for (int t = 0; t < TT; ++t) {
    e_embed_kernel<<<(NB * NN * 32 + 255) / 256, 256, 0, stream>>>(node_emb, e1v, e2v, t, E1b, E2b);

    // A = softmax(relu(E1 @ E2^T))
    adj_logits_wmma<<<(NB * 64 * 64) / 8, 256, 0, stream>>>(E1b, E2b, logits);
    softmax_rows<<<NB * NN, 256, 0, stream>>>(logits, Ab);

    // gate pass: zr = sigmoid(gconv([x,state]))
    build_xg<<<(NB * NN * KPAD + 255) / 256, 256, 0, stream>>>(x_data, todF, dowF, state, zr, 0, t, xg);
    ax_wmma<<<NB * 64, 256, 0, stream>>>(Ab, xg, x1);
    node_gemm_wmma<<<(NN * 8) / 8, 256, 0, stream>>>(xg, x1, WgN, bgN, zr, 128, 0);

    // candidate pass: hc = tanh(gconv([x, z*state]))
    build_xg<<<(NB * NN * KPAD + 255) / 256, 256, 0, stream>>>(x_data, todF, dowF, state, zr, 1, t, xg);
    ax_wmma<<<NB * 64, 256, 0, stream>>>(Ab, xg, x1);
    node_gemm_wmma<<<(NN * 4) / 8, 256, 0, stream>>>(xg, x1, WuN, buN, hc, 64, 1);

    state_update<<<(NB * NN * HID + 255) / 256, 256, 0, stream>>>(zr, hc, state, statesAll, t);
  }

  // ---- temporal self-attention ----
  attention_kernel<<<NB * NN, HID, 0, stream>>>(statesAll, WQ, bQ, WK, bK, WV, bV, out);

  (void)in_sizes; (void)n_in; (void)out_size; (void)ws_size;
}
